// SpareMOE_53953379173318
// MI455X (gfx1250) — matrix operand order repaired and checked by reading, not compiled, not run on hardware
//
#include <hip/hip_runtime.h>
#include <hip/hip_bf16.h>
#include <math.h>

// ---------------------------------------------------------------------------
// Types for CDNA5 WMMA (wave32): bf16 A/B fragments (16 elems), f32 C/D (8).
// ---------------------------------------------------------------------------
typedef __attribute__((ext_vector_type(16))) __bf16 bf16x16;
typedef __attribute__((ext_vector_type(8)))  __bf16 bf16x8;
typedef __attribute__((ext_vector_type(8)))  float  f32x8;

__device__ __forceinline__ unsigned short f2bf(float f) {
    unsigned u = __float_as_uint(f);
    u += 0x7FFFu + ((u >> 16) & 1u);   // round-to-nearest-even truncation
    return (unsigned short)(u >> 16);
}

__device__ __forceinline__ float gelu_exact(float v) {
    return 0.5f * v * (1.0f + erff(v * 0.7071067811865475f));
}

#define TMc 128   // block tile M (8 waves: 4 rows x 2 cols of 32x32)
#define TNc 64    // block tile N
#define TKc 64    // K per LDS stage (2 WMMA K-substeps)

// ---------------------------------------------------------------------------
// Generic bf16-WMMA GEMM: C[M,N] = act( A[M,K] * W[N,K]^T + bias[N] )
//   * A fp32 row-major [M, lda]; W fp32 element (n,k) at
//     Bw[n*ldbN + k*kStride + t]   (kStride=3, t=tap for conv weights).
//   * Conv mode (Lrow>0): rows are (b,l); tap t shifts l by (t-1)*dil with
//     zero padding at batch boundaries (ntaps=3). Plain GEMM: ntaps=1.
//   * epilogue==0 : Cout[m*ldc + n] = val
//   * epilogue==1 : MoE combine: out[b, n, d] += gates[m*numE+e] * val,
//                   m = b*Dd + d, out shaped [B, Hh, Dd] (fused gate-scale +
//                   scatter-add + final transpose).
// 256 threads = 8 waves; double-buffered LDS (one barrier per stage);
// next-stage global_load_b128s issued before the 8 WMMAs of the current
// stage so HBM latency hides under the matrix pipe.
// ---------------------------------------------------------------------------
__global__ __launch_bounds__(256)
void wmma_gemm_kernel(const float* __restrict__ A, int lda, int Lrow,
                      const float* __restrict__ Bw, int ldbN, int kStride,
                      int ntaps, int dil,
                      const float* __restrict__ bias,
                      float* __restrict__ Cout, int ldc,
                      int K, int act, int epilogue,
                      const float* __restrict__ gates, int expertIdx, int numE,
                      int Dd, int Hh)
{
    __shared__ unsigned short As[2][TMc * TKc];   // [m][k] bf16, ping-pong
    __shared__ unsigned short Bs[2][TNc * TKc];   // [n][k] bf16 (= W layout)

    const int tid  = threadIdx.x;
    const int lane = tid & 31;
    const int wave = tid >> 5;
    const int wr   = (wave >> 1) * 32;            // wave row offset (0..96)
    const int wc   = (wave & 1) * 32;             // wave col offset (0/32)
    const int half = (lane >> 4) & 1;
    const int l16  = lane & 15;

    const int blockN = blockIdx.x * TNc;
    const int blockM = blockIdx.y * TMc;

    // staging chunk coordinates (4-element chunks along k)
    int aRow[8], aCol[8];                         // 128x64 / 256 thr
    #pragma unroll
    for (int j = 0; j < 8; ++j) {
        int cidx = j * 256 + tid;
        aRow[j] = cidx >> 4;                      // 16 chunks per row
        aCol[j] = (cidx & 15) * 4;
    }
    int bRow[4], bCol[4];                         // 64x64 / 256 thr
    #pragma unroll
    for (int j = 0; j < 4; ++j) {
        int cidx = j * 256 + tid;
        bRow[j] = cidx >> 4;
        bCol[j] = (cidx & 15) * 4;
    }

    const int kSteps  = K / TKc;
    const int nStages = ntaps * kSteps;

    float4 aReg[8];
    float4 bReg[4];

    // issue global loads for stage si into aReg/bReg
    auto loadStage = [&](int si) {
        int t  = si / kSteps;
        int k0 = (si - t * kSteps) * TKc;
        int shift = (t - (ntaps >> 1)) * dil;     // ntaps=3: (t-1)*dil
        #pragma unroll
        for (int j = 0; j < 8; ++j) {
            int gm = blockM + aRow[j];
            int src;
            if (Lrow > 0) {                       // conv: shifted, padded
                int b = gm / Lrow;
                int l = gm - b * Lrow + shift;
                src = (l >= 0 && l < Lrow) ? (b * Lrow + l) : -1;
            } else {
                src = gm;
            }
            aReg[j] = make_float4(0.f, 0.f, 0.f, 0.f);
            if (src >= 0)
                aReg[j] = *(const float4*)(A + (size_t)src * lda +
                                           (k0 + aCol[j]));
        }
        #pragma unroll
        for (int j = 0; j < 4; ++j) {
            int gn = blockN + bRow[j];
            const float* bp = Bw + (size_t)gn * ldbN +
                              (size_t)(k0 + bCol[j]) * kStride + t;
            if (kStride == 1) {
                bReg[j] = *(const float4*)bp;
            } else {                              // conv weights: stride-3
                bReg[j].x = bp[0];           bReg[j].y = bp[kStride];
                bReg[j].z = bp[2 * kStride]; bReg[j].w = bp[3 * kStride];
            }
        }
    };

    // convert + store staged registers into LDS buffer `buf`
    auto storeStage = [&](int buf) {
        #pragma unroll
        for (int j = 0; j < 8; ++j) {
            ushort4 u = make_ushort4(f2bf(aReg[j].x), f2bf(aReg[j].y),
                                     f2bf(aReg[j].z), f2bf(aReg[j].w));
            *(ushort4*)(&As[buf][aRow[j] * TKc + aCol[j]]) = u;
        }
        #pragma unroll
        for (int j = 0; j < 4; ++j) {
            ushort4 u = make_ushort4(f2bf(bReg[j].x), f2bf(bReg[j].y),
                                     f2bf(bReg[j].z), f2bf(bReg[j].w));
            *(ushort4*)(&Bs[buf][bRow[j] * TKc + bCol[j]]) = u;
        }
    };

    f32x8 acc[2][2] = {{{}, {}}, {{}, {}}};

    // 8 WMMAs on LDS buffer `buf`
    auto compute = [&](int buf) {
        #pragma unroll
        for (int s = 0; s < 2; ++s) {
            const int ko  = s * 32;
            const int klo = half * 8;             // A: K = klo.., 16+klo..
            const int kb  = half * 8;             // B: K = kb.., 16+kb..
            auto ld8 = [](const unsigned short* p) -> bf16x8 {
                return *(const bf16x8*)p;
            };
            bf16x16 a0 = __builtin_shufflevector(
                ld8(&As[buf][(wr + l16) * TKc + ko + klo]),
                ld8(&As[buf][(wr + l16) * TKc + ko + 16 + klo]),
                0,1,2,3,4,5,6,7,8,9,10,11,12,13,14,15);
            bf16x16 a1 = __builtin_shufflevector(
                ld8(&As[buf][(wr + 16 + l16) * TKc + ko + klo]),
                ld8(&As[buf][(wr + 16 + l16) * TKc + ko + 16 + klo]),
                0,1,2,3,4,5,6,7,8,9,10,11,12,13,14,15);
            bf16x16 b0 = __builtin_shufflevector(
                ld8(&Bs[buf][(wc + l16) * TKc + ko + kb]),
                ld8(&Bs[buf][(wc + l16) * TKc + ko + 16 + kb]),
                0,1,2,3,4,5,6,7,8,9,10,11,12,13,14,15);
            bf16x16 b1 = __builtin_shufflevector(
                ld8(&Bs[buf][(wc + 16 + l16) * TKc + ko + kb]),
                ld8(&Bs[buf][(wc + 16 + l16) * TKc + ko + 16 + kb]),
                0,1,2,3,4,5,6,7,8,9,10,11,12,13,14,15);

            acc[0][0] = __builtin_amdgcn_wmma_f32_16x16x32_bf16(
                false, a0, false, b0, (short)0, acc[0][0], false, false);
            acc[0][1] = __builtin_amdgcn_wmma_f32_16x16x32_bf16(
                false, a0, false, b1, (short)0, acc[0][1], false, false);
            acc[1][0] = __builtin_amdgcn_wmma_f32_16x16x32_bf16(
                false, a1, false, b0, (short)0, acc[1][0], false, false);
            acc[1][1] = __builtin_amdgcn_wmma_f32_16x16x32_bf16(
                false, a1, false, b1, (short)0, acc[1][1], false, false);
        }
    };

    // -------- software pipeline: prologue stage 0, then ping-pong --------
    loadStage(0);
    storeStage(0);
    __syncthreads();
    int buf = 0;
    for (int si = 0; si < nStages; ++si) {
        const bool more = (si + 1) < nStages;
        if (more) loadStage(si + 1);   // global loads in flight during WMMAs
        compute(buf);
        if (more) storeStage(buf ^ 1);
        __syncthreads();               // one barrier per stage
        buf ^= 1;
    }

    // ---- epilogue: bias + activation (+ fused MoE combine/transpose) ----
    #pragma unroll
    for (int i = 0; i < 2; ++i) {
        #pragma unroll
        for (int j = 0; j < 2; ++j) {
            const int Mb = blockM + wr + i * 16;
            const int Nb = blockN + wc + j * 16;
            const int nn = Nb + l16;
            const float bv = bias ? bias[nn] : 0.0f;
            #pragma unroll
            for (int r = 0; r < 8; ++r) {
                int   mm = Mb + r + half * 8;
                float v  = acc[i][j][r] + bv;
                if (act == 1) v = gelu_exact(v);
                if (epilogue == 0) {
                    Cout[(size_t)mm * ldc + nn] = v;
                } else {
                    float g  = gates[(size_t)mm * numE + expertIdx];
                    int   bb = mm / Dd, dd = mm - bb * Dd;
                    float* p = Cout + ((size_t)bb * Hh + nn) * Dd + dd;
                    *p += g * v;   // unique owner per launch; experts serial
                }
            }
        }
    }
}

// ---------------------------------------------------------------------------
// Router head: one wave per (b,d) row. 8 dots over L, then sigmoid ->
// softmax -> top-2 -> renormalized gates.
// ---------------------------------------------------------------------------
__global__ __launch_bounds__(256)
void router_head_kernel(const float* __restrict__ h,   // [B, L, D]
                        const float* __restrict__ flw, // [E, L]
                        const float* __restrict__ flb, // [E]
                        float* __restrict__ gates,     // [B*D, E]
                        int Bn, int Ln, int Dn)
{
    int row  = blockIdx.x * (blockDim.x >> 5) + (threadIdx.x >> 5);
    int lane = threadIdx.x & 31;
    if (row >= Bn * Dn) return;
    int b = row / Dn, d = row - b * Dn;

    float acc[8];
    #pragma unroll
    for (int e = 0; e < 8; ++e) acc[e] = 0.0f;
    const float* hb = h + (size_t)b * Ln * Dn + d;
    for (int l = lane; l < Ln; l += 32) {
        float hv = hb[(size_t)l * Dn];
        #pragma unroll
        for (int e = 0; e < 8; ++e) acc[e] += hv * flw[e * Ln + l];
    }
    #pragma unroll
    for (int off = 16; off > 0; off >>= 1) {
        #pragma unroll
        for (int e = 0; e < 8; ++e) acc[e] += __shfl_xor(acc[e], off, 32);
    }
    if (lane == 0) {
        float p[8], m = -1e30f, s = 0.0f;
        #pragma unroll
        for (int e = 0; e < 8; ++e) {
            float sg = 1.0f / (1.0f + expf(-(acc[e] + flb[e])));
            p[e] = sg;
            m = fmaxf(m, sg);
        }
        #pragma unroll
        for (int e = 0; e < 8; ++e) { p[e] = expf(p[e] - m); s += p[e]; }
        float inv_s = 1.0f / s;
        #pragma unroll
        for (int e = 0; e < 8; ++e) p[e] *= inv_s;
        int i1 = 0; float v1 = p[0];
        #pragma unroll
        for (int e = 1; e < 8; ++e) if (p[e] > v1) { v1 = p[e]; i1 = e; }
        int i2 = -1; float v2 = -1.0f;
        #pragma unroll
        for (int e = 0; e < 8; ++e)
            if (e != i1 && p[e] > v2) { v2 = p[e]; i2 = e; }
        float inv = 1.0f / (v1 + v2);
        #pragma unroll
        for (int e = 0; e < 8; ++e)
            gates[(size_t)row * 8 + e] =
                (e == i1) ? v1 * inv : ((e == i2) ? v2 * inv : 0.0f);
    }
}

// Per-expert avg-pool (kernel=stride=f) of x[B,L,D] over L, padded to LMAX.
__global__ __launch_bounds__(256)
void pool_kernel(const float* __restrict__ x, float* __restrict__ xe,
                 int Bn, int Ln, int Dn, int f, int Li, int LMAXc)
{
    int idx = blockIdx.x * blockDim.x + threadIdx.x;
    int total = Bn * Dn * LMAXc;
    if (idx >= total) return;
    int li = idx % LMAXc;
    int m  = idx / LMAXc;
    int d  = m % Dn, b = m / Dn;
    float v = 0.0f;
    if (li < Li) {
        const float* p = x + (size_t)b * Ln * Dn + (size_t)(li * f) * Dn + d;
        for (int j = 0; j < f; ++j) v += p[(size_t)j * Dn];
        v *= (1.0f / (float)f);
    }
    xe[idx] = v;
}

__global__ __launch_bounds__(256)
void zero_kernel(float* __restrict__ p, size_t n)
{
    size_t i = (size_t)blockIdx.x * blockDim.x + threadIdx.x;
    size_t stride = (size_t)gridDim.x * blockDim.x;
    for (; i < n; i += stride) p[i] = 0.0f;
}

// ---------------------------------------------------------------------------
extern "C" void kernel_launch(void* const* d_in, const int* in_sizes, int n_in,
                              void* d_out, int out_size, void* d_ws, size_t ws_size,
                              hipStream_t stream)
{
    const float* x     = (const float*)d_in[0];   // [B, L, D]
    const float* convw = (const float*)d_in[1];   // [3, D, D, 3]
    const float* convb = (const float*)d_in[2];   // [3, D]
    const float* flw   = (const float*)d_in[3];   // [E, L]
    const float* flb   = (const float*)d_in[4];   // [E]
    const float* w1    = (const float*)d_in[5];   // [E, H, LMAX]
    const float* b1    = (const float*)d_in[6];   // [E, H]
    const float* w2    = (const float*)d_in[7];   // [E, H, H]
    const float* b2    = (const float*)d_in[8];   // [E, H]
    const float* w3    = (const float*)d_in[9];   // [E, H, H]
    const float* b3    = (const float*)d_in[10];  // [E, H]
    float* out = (float*)d_out;                   // [B, H, D]

    const int B = 8, L = 1024, D = 256, E = 8, H = 1024, LMAX = 512;

    // workspace carve (floats): ~36 MB total
    float* hA    = (float*)d_ws;                       // [B*L, D]
    float* hB    = hA    + (size_t)B * L * D;          // [B*L, D]
    float* gates = hB    + (size_t)B * L * D;          // [B*D, E]
    float* xe    = gates + (size_t)B * D * E;          // [B*D, LMAX]
    float* h1    = xe    + (size_t)B * D * LMAX;       // [B*D, H]
    float* h2    = h1    + (size_t)B * D * H;          // [B*D, H]
    (void)ws_size; (void)in_sizes; (void)n_in; (void)out_size;

    dim3 blk(256);

    // --- router: 3 dilated D->D convs as 3-tap WMMA GEMMs, gelu fused ---
    dim3 cgrid(D / TNc, (B * L) / TMc);
    wmma_gemm_kernel<<<cgrid, blk, 0, stream>>>(
        x,  D, L, convw + 0 * (size_t)D * D * 3, D * 3, 3, 3, 1,
        convb + 0 * D, hA, D, D, /*act*/1, /*epi*/0, nullptr, 0, E, D, H);
    wmma_gemm_kernel<<<cgrid, blk, 0, stream>>>(
        hA, D, L, convw + 1 * (size_t)D * D * 3, D * 3, 3, 3, 2,
        convb + 1 * D, hB, D, D, 1, 0, nullptr, 0, E, D, H);
    wmma_gemm_kernel<<<cgrid, blk, 0, stream>>>(
        hB, D, L, convw + 2 * (size_t)D * D * 3, D * 3, 3, 3, 4,
        convb + 2 * D, hA, D, D, 1, 0, nullptr, 0, E, D, H);

    // --- gates ---
    router_head_kernel<<<(B * D) / 8, 256, 0, stream>>>(hA, flw, flb, gates,
                                                        B, L, D);

    // --- output accumulator ---
    zero_kernel<<<2048, 256, 0, stream>>>(out, (size_t)B * H * D);

    // --- experts: pool -> 3 WMMA GEMMs; layer 3 fuses gate*val scatter-add ---
    dim3 egrid(H / TNc, (B * D) / TMc);
    for (int e = 0; e < E; ++e) {
        int f = e + 2, Li = L / f;
        pool_kernel<<<(B * D * LMAX + 255) / 256, 256, 0, stream>>>(
            x, xe, B, L, D, f, Li, LMAX);
        wmma_gemm_kernel<<<egrid, blk, 0, stream>>>(
            xe, LMAX, 0, w1 + (size_t)e * H * LMAX, LMAX, 1, 1, 0,
            b1 + (size_t)e * H, h1, H, LMAX, /*act*/1, /*epi*/0,
            nullptr, 0, E, D, H);
        wmma_gemm_kernel<<<egrid, blk, 0, stream>>>(
            h1, H, 0, w2 + (size_t)e * H * H, H, 1, 1, 0,
            b2 + (size_t)e * H, h2, H, H, 1, 0, nullptr, 0, E, D, H);
        wmma_gemm_kernel<<<egrid, blk, 0, stream>>>(
            h2, H, 0, w3 + (size_t)e * H * H, H, 1, 1, 0,
            b3 + (size_t)e * H, out, 0, H, /*act*/0, /*epi combine*/1,
            gates, e, E, D, H);
    }
}
